// SAGEEncoder_83803401879709
// MI455X (gfx1250) — matrix operand order repaired and checked by reading, not compiled, not run on hardware
//
#include <hip/hip_runtime.h>

// ---------------------------------------------------------------------------
// SAGEEncoder for MI455X (gfx1250): edge-scatter (HBM/atomic bound) +
// bf16 WMMA GEMMs (v_wmma_f32_16x16x32_bf16) with fused bias+ReLU epilogue.
// ---------------------------------------------------------------------------

typedef __attribute__((ext_vector_type(16))) __bf16 v16bf;
typedef __attribute__((ext_vector_type(8)))  __bf16 v8bf;
typedef __attribute__((ext_vector_type(8)))  float  v8f;

#define DIMV 128   // IN_DIM == HID == 128 for both layers

// ---------------- utility kernels ------------------------------------------

__global__ void k_zero_f32(float* __restrict__ p, int n) {
  int i = blockIdx.x * blockDim.x + threadIdx.x;
  if (i < n) p[i] = 0.0f;
}

__global__ void k_cast_bf16(const float* __restrict__ in, __bf16* __restrict__ out, int n) {
  int i = blockIdx.x * blockDim.x + threadIdx.x;
  if (i < n) out[i] = (__bf16)in[i];
}

__global__ void k_degree(const int* __restrict__ dst, float* __restrict__ deg, int E) {
  int e = blockIdx.x * blockDim.x + threadIdx.x;
  if (e < E) atomicAdd(&deg[dst[e]], 1.0f);
}

__global__ void k_recip(const float* __restrict__ deg, float* __restrict__ rdeg, int n) {
  int i = blockIdx.x * blockDim.x + threadIdx.x;
  if (i < n) rdeg[i] = 1.0f / fmaxf(deg[i], 1.0f);
}

// one wave (32 lanes) per edge, each lane owns 4 contiguous dims
__global__ void k_scatter_add(const float* __restrict__ feat, const int* __restrict__ src,
                              const int* __restrict__ dst, float* __restrict__ agg, int E) {
  long t = (long)blockIdx.x * blockDim.x + threadIdx.x;
  int e = (int)(t >> 5);
  if (e >= E) return;
  int lane = (int)(t & 31);
  int s = src[e];   // wave-uniform -> scalarized
  int d = dst[e];
  const float4 v = *(const float4*)(feat + (size_t)s * DIMV + lane * 4);
  float* o = agg + (size_t)d * DIMV + lane * 4;
  atomicAdd(o + 0, v.x);
  atomicAdd(o + 1, v.y);
  atomicAdd(o + 2, v.z);
  atomicAdd(o + 3, v.w);
}

// agg (f32) * rdeg[node] -> bf16 (mean aggregation, converted for WMMA)
__global__ void k_norm_cast(const float* __restrict__ agg, const float* __restrict__ rdeg,
                            __bf16* __restrict__ out, int total) {
  int i = blockIdx.x * blockDim.x + threadIdx.x;
  if (i < total) out[i] = (__bf16)(agg[i] * rdeg[i >> 7]);   // DIMV == 128
}

// ---------------- fused SAGE GEMM ------------------------------------------
// out[m,n] = relu( sum_k agg[m,k]*Wl[n,k] + b[n] + sum_k x[m,k]*Wr[n,k] )
// A fragment: 16-bit A layout — lanes 0-15 row m, half selects K phase:
//   elems 0..7  -> K = k0 + 8*half + (0..7)
//   elems 8..15 -> K = k0 + 8*half + 16 + (0..7)
__device__ __forceinline__ v16bf load_fragA(const __bf16* __restrict__ base,
                                            int row, int k0, int half) {
  const __bf16* p = base + (size_t)row * DIMV + k0 + half * 8;
  v8bf lo = *(const v8bf*)(p);        // 16B aligned
  v8bf hi = *(const v8bf*)(p + 16);   // 16B aligned
  return __builtin_shufflevector(lo, hi, 0, 1, 2, 3, 4, 5, 6, 7,
                                         8, 9, 10, 11, 12, 13, 14, 15);
}

__global__ __launch_bounds__(256)
void k_sage_gemm(const __bf16* __restrict__ aggbf,  // [N,128] mean-aggregated
                 const __bf16* __restrict__ xbf,    // [N,128] self features
                 const __bf16* __restrict__ Wl,     // [128,128] row-major
                 const __bf16* __restrict__ Wr,     // [128,128] row-major
                 const float*  __restrict__ bias,   // [128]
                 float* __restrict__ out,           // [N,128]
                 int nNodes) {
  const int wave = threadIdx.x >> 5;      // 0..7 -> output-column tile
  const int lane = threadIdx.x & 31;
  const int half = lane >> 4;             // 0 | 1
  const int nl   = lane & 15;
  const int row0 = blockIdx.x * 16;
  const int n0   = wave * 16;

  int arow = row0 + nl;
  if (arow >= nNodes) arow = nNodes - 1;  // tail-safe (N=50000 is exact anyway)

  v8f c = {};
#pragma unroll
  for (int k0 = 0; k0 < DIMV; k0 += 32) {
    v16bf aAgg = load_fragA(aggbf, arow, k0, half);
    v16bf aX   = load_fragA(xbf,   arow, k0, half);
    // B[k,n] = W[n,k]: lane's column n -> K = k0 + 8*half + (0..7), then +16 (same as A)
    v16bf bL = load_fragA(Wl, n0 + nl, k0, half);
    v16bf bR = load_fragA(Wr, n0 + nl, k0, half);
    c = __builtin_amdgcn_wmma_f32_16x16x32_bf16(false, aAgg, false, bL, (short)0, c, false, false);
    c = __builtin_amdgcn_wmma_f32_16x16x32_bf16(false, aX,   false, bR, (short)0, c, false, false);
  }

  const float bv = bias[n0 + nl];
#pragma unroll
  for (int r = 0; r < 8; ++r) {
    int m = row0 + r + half * 8;          // C/D layout: VGPR r -> M = r + 8*half
    if (m < nNodes) {
      float v = c[r] + bv;
      out[(size_t)m * DIMV + n0 + nl] = v > 0.0f ? v : 0.0f;
    }
  }
}

// ---------------- host-side orchestration ----------------------------------

extern "C" void kernel_launch(void* const* d_in, const int* in_sizes, int n_in,
                              void* d_out, int out_size, void* d_ws, size_t ws_size,
                              hipStream_t stream) {
  const float* x   = (const float*)d_in[0];
  const int*   ei  = (const int*)d_in[1];
  const float* W1l = (const float*)d_in[2];
  const float* b1  = (const float*)d_in[3];
  const float* W1r = (const float*)d_in[4];
  const float* W2l = (const float*)d_in[5];
  const float* b2  = (const float*)d_in[6];
  const float* W2r = (const float*)d_in[7];
  float* out = (float*)d_out;

  const int N = in_sizes[0] / DIMV;
  const int E = in_sizes[1] / 2;
  const int* src = ei;
  const int* dst = ei + E;
  const int nd = N * DIMV;                // 6.4M elements

  // workspace carve-out (256B aligned slices)
  char* ws = (char*)d_ws;
  size_t off = 0;
  auto alloc = [&](size_t bytes) -> char* {
    char* p = ws + off;
    off = (off + bytes + 255) & ~(size_t)255;
    return p;
  };
  __bf16* xbf   = (__bf16*)alloc((size_t)nd * 2);
  __bf16* hbf   = (__bf16*)alloc((size_t)nd * 2);
  __bf16* aggbf = (__bf16*)alloc((size_t)nd * 2);
  float*  aggf  = (float*) alloc((size_t)nd * 4);
  float*  hf    = (float*) alloc((size_t)nd * 4);
  float*  deg   = (float*) alloc((size_t)N * 4);
  float*  rdeg  = (float*) alloc((size_t)N * 4);
  __bf16* wl1   = (__bf16*)alloc((size_t)DIMV * DIMV * 2);
  __bf16* wr1   = (__bf16*)alloc((size_t)DIMV * DIMV * 2);
  __bf16* wl2   = (__bf16*)alloc((size_t)DIMV * DIMV * 2);
  __bf16* wr2   = (__bf16*)alloc((size_t)DIMV * DIMV * 2);

  const int BT = 256;
  const int nbElem = (nd + BT - 1) / BT;       // per-element passes
  const int nbNode = (N + BT - 1) / BT;
  const int nbEdge = (E + BT - 1) / BT;
  const int nbScat = (int)(((long)E * 32 + BT - 1) / BT);
  const int nbW    = (DIMV * DIMV + BT - 1) / BT;
  const int nTiles = (N + 15) / 16;

  // degree (shared by both layers)
  k_zero_f32 <<<nbNode, BT, 0, stream>>>(deg, N);
  k_degree   <<<nbEdge, BT, 0, stream>>>(dst, deg, E);
  k_recip    <<<nbNode, BT, 0, stream>>>(deg, rdeg, N);

  // one-time bf16 conversions
  k_cast_bf16<<<nbElem, BT, 0, stream>>>(x,   xbf, nd);
  k_cast_bf16<<<nbW,    BT, 0, stream>>>(W1l, wl1, DIMV * DIMV);
  k_cast_bf16<<<nbW,    BT, 0, stream>>>(W1r, wr1, DIMV * DIMV);
  k_cast_bf16<<<nbW,    BT, 0, stream>>>(W2l, wl2, DIMV * DIMV);
  k_cast_bf16<<<nbW,    BT, 0, stream>>>(W2r, wr2, DIMV * DIMV);

  // ---- layer 1: h = relu(mean(x[src]->dst) @ W1l^T + b1 + x @ W1r^T) ----
  k_zero_f32    <<<nbElem, BT, 0, stream>>>(aggf, nd);
  k_scatter_add <<<nbScat, BT, 0, stream>>>(x, src, dst, aggf, E);
  k_norm_cast   <<<nbElem, BT, 0, stream>>>(aggf, rdeg, aggbf, nd);
  k_sage_gemm   <<<nTiles, BT, 0, stream>>>(aggbf, xbf, wl1, wr1, b1, hf, N);

  // ---- layer 2: out = relu(mean(h[src]->dst) @ W2l^T + b2 + h @ W2r^T) ----
  k_cast_bf16   <<<nbElem, BT, 0, stream>>>(hf, hbf, nd);
  k_zero_f32    <<<nbElem, BT, 0, stream>>>(aggf, nd);
  k_scatter_add <<<nbScat, BT, 0, stream>>>(hf, src, dst, aggf, E);
  k_norm_cast   <<<nbElem, BT, 0, stream>>>(aggf, rdeg, aggbf, nd);
  k_sage_gemm   <<<nTiles, BT, 0, stream>>>(aggbf, hbf, wl2, wr2, b2, out, N);
}
